// DifferentiableTokenSelection_30691836297739
// MI455X (gfx1250) — compile-verified, matrix operand order checked by reading, not run on hardware
//
#include <hip/hip_runtime.h>

// ---------------------------------------------------------------------------
// DifferentiableTokenSelection on MI455X (gfx1250, wave32, WMMA bf16 16x16x32)
//   b=2, t*n=M=65536, e=E=512, top_k=K=256
//   prep : zero out/Z, convert W -> bf16 (Wb)
//   pass1: P^T = exp(X @ W^T)  (bf16, stored K-major), Z[k] = sum_m P[m][k]
//          W tile double-buffered in LDS via global_load_async_to_lds_b128
//   pass2: out += P^T @ X   (split-M, f32 atomics)
//   norm : out /= Z
// bias cancels in softmax over the token axis -> intentionally unused.
// ---------------------------------------------------------------------------

typedef __attribute__((ext_vector_type(16))) __bf16          v16bf;
typedef __attribute__((ext_vector_type(2)))  __bf16          v2bf;
typedef __attribute__((ext_vector_type(2)))  float           v2f;
typedef __attribute__((ext_vector_type(8)))  float           v8f;
typedef __attribute__((ext_vector_type(4)))  float           v4f;
typedef __attribute__((ext_vector_type(8)))  unsigned short  v8u;
typedef __attribute__((ext_vector_type(2)))  unsigned int    v2u;

union Frag32 { v8u h[2]; v16bf v; };   // 32 bytes: one 16-bit WMMA operand

#define MTOK   65536
#define EDIM   512
#define KDIM   256
#define NBATCH 2
#define LPITCH 40        // LDS row pitch in ushorts (80B) -> conflict-free b128 reads

// gfx1250 async global->LDS copy path (ASYNCcnt); set to 0 to fall back.
#define USE_ASYNC_LDS 1

// ---- f32 -> bf16 (RNE), packed pair -> one 32-bit word --------------------
// __builtin_convertvector lets the backend select v_cvt_pk_bf16_f32 on gfx1250.
__device__ __forceinline__ unsigned int cvt2u(float lo, float hi) {
  v2f f; f[0] = lo; f[1] = hi;
  v2bf t = __builtin_convertvector(f, v2bf);
  return __builtin_bit_cast(unsigned int, t);
}

__device__ __forceinline__ v8u pack8(v4f a, v4f b) {
  union { unsigned int w[4]; v8u u; } r;
  r.w[0] = cvt2u(a.x, a.y);
  r.w[1] = cvt2u(a.z, a.w);
  r.w[2] = cvt2u(b.x, b.y);
  r.w[3] = cvt2u(b.z, b.w);
  return r.u;
}

// ---------------------------------------------------------------------------
// Prep: zero the f32 accumulator (d_out) and Z; convert W to bf16 once.
// Grid: 1024 x 256.
// ---------------------------------------------------------------------------
__global__ void k_prep(const float* __restrict__ W, float* __restrict__ out,
                       float* __restrict__ Z, unsigned short* __restrict__ Wb) {
  int i = blockIdx.x * 256 + threadIdx.x;
  if (i < NBATCH * KDIM * EDIM) out[i] = 0.0f;
  if (i < NBATCH * KDIM)        Z[i]   = 0.0f;
  if (i < (KDIM * EDIM) / 4) {
    v4f w4 = *(const v4f*)(W + (size_t)i * 4);
    v2u p; p[0] = cvt2u(w4.x, w4.y); p[1] = cvt2u(w4.z, w4.w);
    *(v2u*)(Wb + (size_t)i * 4) = p;
  }
}

// ---------------------------------------------------------------------------
// Pass 1: scores + exp + transpose-store + column sums.
// Grid: (M/128, NBATCH). Block: 256 threads = 8 waves.
// Wave w owns rows m = mblock + 16*w .. +15, and ALL 256 k columns (16 tiles).
// ---------------------------------------------------------------------------
__global__ __launch_bounds__(256) void k_pass1(const float* __restrict__ X,
                                               const unsigned short* __restrict__ Wb,
                                               unsigned short* __restrict__ PT,
                                               float* __restrict__ Z) {
  __shared__ __attribute__((aligned(16))) unsigned short wtile[2][KDIM][LPITCH];
  __shared__ float zsh[KDIM];

  const int tid    = threadIdx.x;
  const int lane   = tid & 31;
  const int wave   = tid >> 5;
  const int mblock = blockIdx.x * 128;
  const int batch  = blockIdx.y;
  const int mwave  = mblock + wave * 16;
  const size_t xbase = (size_t)batch * MTOK * EDIM;

  zsh[tid] = 0.0f;     // blockDim == KDIM

  v8f acc[16];
#pragma unroll
  for (int t = 0; t < 16; ++t)
#pragma unroll
    for (int r = 0; r < 8; ++r) acc[t][r] = 0.0f;

  const int   mrow = mwave + (lane & 15);
  const int   b0   = (lane >> 4) << 3;           // 0 or 8: A/B element chunk base
  const float* xrow = X + xbase + (size_t)mrow * EDIM;

  // stage pre-converted bf16 W[k=0..255][e0..e0+31] into LDS buffer `buf`
  auto stageW = [&](int e0, int buf) {
#pragma unroll
    for (int s = 0; s < 4; ++s) {
      int j   = tid + (s << 8);                  // b128 chunk id 0..1023
      int row = j >> 2;                          // k row 0..255
      int c8  = (j & 3) << 3;                    // 0,8,16,24
      const unsigned short* gp = Wb + (size_t)row * EDIM + e0 + c8;
      unsigned short* lp = &wtile[buf][row][c8];
#if USE_ASYNC_LDS
      unsigned int ldsoff = (unsigned int)(uintptr_t)lp;   // LDS byte offset
      asm volatile("global_load_async_to_lds_b128 %0, %1, off"
                   :: "v"(ldsoff), "v"(gp) : "memory");
#else
      *(v8u*)lp = *(const v8u*)gp;
#endif
    }
  };

  stageW(0, 0);
#if USE_ASYNC_LDS
  asm volatile("s_wait_asynccnt 0x0" ::: "memory");
#endif
  __syncthreads();

  int cur = 0;
#pragma unroll 1
  for (int e0 = 0; e0 < EDIM; e0 += 32) {
    // prefetch next W tile into the other buffer while we compute on `cur`
    if (e0 + 32 < EDIM) stageW(e0 + 32, cur ^ 1);

    // A fragment: X[m][e0+b0..+7], X[m][e0+b0+16..+23] (ISA 16-bit A layout)
    const float* xp = xrow + e0 + b0;
    Frag32 af;
    af.h[0] = pack8(*(const v4f*)(xp),      *(const v4f*)(xp + 4));
    af.h[1] = pack8(*(const v4f*)(xp + 16), *(const v4f*)(xp + 20));

#pragma unroll
    for (int kt = 0; kt < 16; ++kt) {
      const int krow = (kt << 4) + (lane & 15); // B lane carries N=k
      Frag32 bf;
      bf.h[0] = *(const v8u*)&wtile[cur][krow][b0];
      bf.h[1] = *(const v8u*)&wtile[cur][krow][b0 + 16];
      acc[kt] = __builtin_amdgcn_wmma_f32_16x16x32_bf16(
          false, af.v, false, bf.v, (short)0, acc[kt], false, false);
    }

#if USE_ASYNC_LDS
    asm volatile("s_wait_asynccnt 0x0" ::: "memory");   // own async copies landed
#endif
    __syncthreads();                                    // all waves done with `cur`
    cur ^= 1;
  }

  // epilogue: exp, transpose-store bf16 P^T, per-column partial sums
  const size_t mstore = (size_t)mwave + ((lane >> 4) << 3);
#pragma unroll
  for (int kt = 0; kt < 16; ++kt) {
    const int k = (kt << 4) + (lane & 15);
    union { unsigned int w[4]; v8u u; } p;
    float zsum = 0.0f;
#pragma unroll
    for (int rr = 0; rr < 4; ++rr) {
      float ev0 = __expf(acc[kt][2 * rr + 0]);   // TAU == 1
      float ev1 = __expf(acc[kt][2 * rr + 1]);
      p.w[rr] = cvt2u(ev0, ev1);
      zsum += ev0 + ev1;
    }
    *(v8u*)(PT + ((size_t)(batch * KDIM + k)) * MTOK + mstore) = p.u;  // 16B store
    atomicAdd(&zsh[k], zsum);                    // ds_add_f32
  }
  __syncthreads();
  atomicAdd(&Z[batch * KDIM + tid], zsh[tid]);   // one global atomic per column
}

// ---------------------------------------------------------------------------
// Pass 2: out += P^T @ X  (split over M for occupancy).
// Grid: (16 m-splits, 4 e-blocks, batch*2 + k-block). Block: 256 = 8 waves.
// Wave w owns k rows kw..kw+15 and 128 e columns (8 tiles).
// ---------------------------------------------------------------------------
__global__ __launch_bounds__(256) void k_pass2(const float* __restrict__ X,
                                               const unsigned short* __restrict__ PT,
                                               float* __restrict__ out) {
  __shared__ __attribute__((aligned(16))) unsigned short xt[128][LPITCH];

  const int tid    = threadIdx.x;
  const int lane   = tid & 31;
  const int wave   = tid >> 5;
  const int msplit = blockIdx.x;
  const int ebase  = blockIdx.y * 128;
  const int kb     = blockIdx.z & 1;
  const int batch  = blockIdx.z >> 1;
  const int kw     = kb * 128 + wave * 16;
  const size_t xbase = (size_t)batch * MTOK * EDIM;

  v8f acc[8];
#pragma unroll
  for (int t = 0; t < 8; ++t)
#pragma unroll
    for (int r = 0; r < 8; ++r) acc[t][r] = 0.0f;

  const int b0   = (lane >> 4) << 3;
  const int krow = kw + (lane & 15);
  const unsigned short* prow = PT + ((size_t)(batch * KDIM + krow)) * MTOK;

#pragma unroll 1
  for (int mi = 0; mi < 4096; mi += 32) {
    const int mbase = msplit * 4096 + mi;
    __syncthreads();
    // stage X[mbase..+31][ebase..+127] -> bf16, TRANSPOSED in LDS: xt[e][m].
    // Each thread handles an adjacent m-pair at 4 e-columns, so one
    // cvt2u(row_m, row_m+1) == one packed 32-bit LDS store.
#pragma unroll
    for (int s = 0; s < 2; ++s) {
      int mrp = (tid >> 5) + (s << 3);           // m-pair index 0..15
      int c4  = (tid & 31) << 2;                 // e column 0..124
      const float* xp0 = X + xbase + (size_t)(mbase + 2 * mrp) * EDIM + ebase + c4;
      v4f xa = *(const v4f*)(xp0);
      v4f xb = *(const v4f*)(xp0 + EDIM);
      *(unsigned int*)&xt[c4 + 0][2 * mrp] = cvt2u(xa.x, xb.x);
      *(unsigned int*)&xt[c4 + 1][2 * mrp] = cvt2u(xa.y, xb.y);
      *(unsigned int*)&xt[c4 + 2][2 * mrp] = cvt2u(xa.z, xb.z);
      *(unsigned int*)&xt[c4 + 3][2 * mrp] = cvt2u(xa.w, xb.w);
    }
    __syncthreads();

    // A fragment from P^T row (contiguous bf16, 2 x b128)
    Frag32 af;
    af.h[0] = *(const v8u*)(prow + mbase + b0);
    af.h[1] = *(const v8u*)(prow + mbase + b0 + 16);

#pragma unroll
    for (int te = 0; te < 8; ++te) {
      const int ecol = (te << 4) + (lane & 15);  // B lane carries N=e
      Frag32 bf;
      bf.h[0] = *(const v8u*)&xt[ecol][b0];      // ds_load_b128
      bf.h[1] = *(const v8u*)&xt[ecol][b0 + 16];
      acc[te] = __builtin_amdgcn_wmma_f32_16x16x32_bf16(
          false, af.v, false, bf.v, (short)0, acc[te], false, false);
    }
  }

  // accumulate partial tiles into out (f32 atomics; normalized later)
#pragma unroll
  for (int te = 0; te < 8; ++te) {
    const int ecol = ebase + (te << 4) + (lane & 15);
#pragma unroll
    for (int r = 0; r < 8; ++r) {
      const int kout = kw + r + ((lane >> 4) << 3);   // C layout: M = r + 8*(lane/16)
      atomicAdd(&out[((size_t)(batch * KDIM + kout)) * EDIM + ecol], acc[te][r]);
    }
  }
}

// ---------------------------------------------------------------------------
__global__ void k_norm(float* __restrict__ out, const float* __restrict__ Z) {
  int i = blockIdx.x * 256 + threadIdx.x;     // < NBATCH*KDIM*EDIM
  int k     = (i >> 9) & (KDIM - 1);          // EDIM == 512
  int batch = i >> 17;                        // / (KDIM*EDIM)
  out[i] = out[i] / Z[batch * KDIM + k];
}

// ---------------------------------------------------------------------------
extern "C" void kernel_launch(void* const* d_in, const int* in_sizes, int n_in,
                              void* d_out, int out_size, void* d_ws, size_t ws_size,
                              hipStream_t stream) {
  (void)in_sizes; (void)n_in; (void)out_size; (void)ws_size;
  const float* X = (const float*)d_in[0];
  const float* W = (const float*)d_in[1];
  // d_in[2] (bias) is constant along the softmax/token axis -> cancels; unused.
  float* out = (float*)d_out;

  // workspace layout
  unsigned short* PT = (unsigned short*)d_ws;                 // 64 MiB: bf16 exp(S)^T
  char* wsp = (char*)d_ws + (size_t)NBATCH * KDIM * MTOK * sizeof(unsigned short);
  float* Z = (float*)wsp;                                     // 2 KiB
  unsigned short* Wb = (unsigned short*)(wsp + 4096);         // 256 KiB bf16 W

  k_prep <<<dim3(1024),               dim3(256), 0, stream>>>(W, out, Z, Wb);
  k_pass1<<<dim3(MTOK / 128, NBATCH), dim3(256), 0, stream>>>(X, Wb, PT, Z);
  k_pass2<<<dim3(16, 4, 4),           dim3(256), 0, stream>>>(X, PT, out);
  k_norm <<<dim3(1024),               dim3(256), 0, stream>>>(out, Z);
}
